// AttentionBlock_57947698757862
// MI455X (gfx1250) — matrix operand order repaired and checked by reading, not compiled, not run on hardware
//
#include <hip/hip_runtime.h>
#include <hip/hip_bf16.h>

typedef __bf16 bf16_t;
typedef __attribute__((ext_vector_type(16))) __bf16 v16bf;
typedef __attribute__((ext_vector_type(8)))  __bf16 v8bf;
typedef __attribute__((ext_vector_type(8)))  float  v8f;

#define WMMA_BF16(a, b, c) \
  __builtin_amdgcn_wmma_f32_16x16x32_bf16(false, (a), false, (b), (short)0, (c), false, false)

// Async global->LDS copy of 32 contiguous bytes (2 x B128), tracked by ASYNCcnt.
// LDS address = low 32 bits of the generic pointer (ISA 10.2 aperture rule).
__device__ __forceinline__ void stage_async32(const bf16_t* g, bf16_t* lds) {
  unsigned la = (unsigned)(size_t)(void*)lds;
  asm volatile("global_load_async_to_lds_b128 %0, %1, off\n\t"
               "global_load_async_to_lds_b128 %0, %1, off offset:16"
               :: "v"(la), "v"(g) : "memory");
}
__device__ __forceinline__ void wait_async0() {
  asm volatile("s_wait_asynccnt 0x0" ::: "memory");
}

// ---------------------------------------------------------------------------
// f32 -> bf16 conversion
// ---------------------------------------------------------------------------
__global__ void k_cvt_bf16(const float* __restrict__ src, bf16_t* __restrict__ dst, int n) {
  int i = blockIdx.x * blockDim.x + threadIdx.x;
  int stride = gridDim.x * blockDim.x;
  for (; i < n; i += stride) dst[i] = (bf16_t)src[i];
}

// ---------------------------------------------------------------------------
// f32 [K][N] -> bf16 transposed [N][K]  (32x32 LDS tile, 256 threads)
// ---------------------------------------------------------------------------
__launch_bounds__(256)
__global__ void k_cvt_t_bf16(const float* __restrict__ src, bf16_t* __restrict__ dst,
                             int K, int N) {
  __shared__ float tile[32][33];
  const int kt = blockIdx.y * 32;
  const int nt = blockIdx.x * 32;
  const int tx = threadIdx.x & 31;
  const int ty = threadIdx.x >> 5;  // 0..7
  #pragma unroll
  for (int r = 0; r < 32; r += 8)
    tile[ty + r][tx] = src[(long long)(kt + ty + r) * N + nt + tx];
  __syncthreads();
  #pragma unroll
  for (int r = 0; r < 32; r += 8)
    dst[(long long)(nt + ty + r) * K + kt + tx] = (bf16_t)tile[tx][ty + r];
}

// ---------------------------------------------------------------------------
// bf16 WMMA GEMM: C = act(alpha * A @ B^T + bias)
//   A: [M,K] row-major bf16;  B: [N,K] row-major bf16 (always "transB" layout)
//   block = 256 threads = 8 waves (2 M-waves x 4 N-waves), tile 128x128, K-step 32
//   Ping-pong LDS buffers filled with async global->LDS B128 copies.
//   Outputs: Cf (f32 [M,N]), Cb (bf16 [M,N]), Ct (bf16 transposed: Ct[n*ldct+m])
// ---------------------------------------------------------------------------
__launch_bounds__(256)
__global__ void k_gemm_wmma(const bf16_t* __restrict__ A, long long strideA,
                            const bf16_t* __restrict__ B, long long strideB,
                            const float* __restrict__ bias,
                            float* __restrict__ Cf, bf16_t* __restrict__ Cb, long long strideC,
                            bf16_t* __restrict__ Ct, long long strideCt, int ldct,
                            int K, int lda, int ldb, int ldc,
                            float alpha, int gelu) {
  __shared__ bf16_t As[2][128 * 32];
  __shared__ bf16_t Bs[2][128 * 32];

  const int t    = threadIdx.x;
  const int lane = t & 31;
  const int wv   = t >> 5;
  const int wm   = (wv >> 2) * 64;  // wave M origin in tile (0 or 64)
  const int wn   = (wv & 3) * 32;   // wave N origin in tile (0..96)

  const long long z = blockIdx.z;
  A += z * strideA;
  B += z * strideB;
  const long long cOff  = z * strideC;
  const long long ctOff = z * strideCt;

  const int m0 = blockIdx.y * 128;
  const int n0 = blockIdx.x * 128;

  v8f acc[4][2];
  #pragma unroll
  for (int i = 0; i < 4; ++i)
    #pragma unroll
    for (int j = 0; j < 2; ++j)
      #pragma unroll
      for (int e = 0; e < 8; ++e) acc[i][j][e] = 0.0f;

  const int sr = t >> 1;         // staging row 0..127
  const int sh = (t & 1) * 16;   // staging k-half 0 or 16
  const bf16_t* aP = A + (long long)(m0 + sr) * lda + sh;  // + k0 per step
  const bf16_t* bP = B + (long long)(n0 + sr) * ldb + sh;
  const int sOff = sr * 32 + sh;

  // prologue: stage k0 = 0 into buffer 0
  stage_async32(aP, &As[0][sOff]);
  stage_async32(bP, &Bs[0][sOff]);

  const int rsel = lane & 15;
  const int hsel = lane >> 4;
  int p = 0;
  for (int k0 = 0; k0 < K; k0 += 32) {
    wait_async0();      // our async writes to buffer p are in LDS
    __syncthreads();    // everyone's writes landed; everyone done reading p^1

    if (k0 + 32 < K) {  // prefetch next slab into the other buffer (overlaps WMMA)
      stage_async32(aP + k0 + 32, &As[p ^ 1][sOff]);
      stage_async32(bP + k0 + 32, &Bs[p ^ 1][sOff]);
    }

    // ---- fragments per ISA wave32 layouts ----
    v16bf afr[4];
    #pragma unroll
    for (int i = 0; i < 4; ++i) {
      // A: lane<16 -> K {0..7,16..23}; lane>=16 -> K {8..15,24..31}
      const bf16_t* pa = &As[p][(wm + i * 16 + rsel) * 32 + hsel * 8];
      v8bf lo = *(const v8bf*)pa;
      v8bf hi = *(const v8bf*)(pa + 16);
      #pragma unroll
      for (int e = 0; e < 8; ++e) { afr[i][e] = lo[e]; afr[i][e + 8] = hi[e]; }
    }
    v16bf bfr[2];
    #pragma unroll
    for (int j = 0; j < 2; ++j) {
      // B: lane<16 -> K {0..7,16..23}; lane>=16 -> K {8..15,24..31}
      const bf16_t* pb = &Bs[p][(wn + j * 16 + rsel) * 32 + hsel * 8];
      v8bf lo = *(const v8bf*)pb;
      v8bf hi = *(const v8bf*)(pb + 16);
      #pragma unroll
      for (int e = 0; e < 8; ++e) { bfr[j][e] = lo[e]; bfr[j][e + 8] = hi[e]; }
    }
    #pragma unroll
    for (int i = 0; i < 4; ++i)
      #pragma unroll
      for (int j = 0; j < 2; ++j)
        acc[i][j] = WMMA_BF16(afr[i], bfr[j], acc[i][j]);

    p ^= 1;
  }

  // ---- epilogue: C layout VGPR e -> M=e (+8 for lanes 16..31), lane -> N ----
  const int rb = m0 + wm + ((lane >> 4) * 8);
  const int cb = n0 + wn + rsel;
  #pragma unroll
  for (int i = 0; i < 4; ++i) {
    #pragma unroll
    for (int j = 0; j < 2; ++j) {
      const int col = cb + j * 16;
      const float bsv = bias ? bias[col] : 0.0f;
      float vals[8];
      #pragma unroll
      for (int e = 0; e < 8; ++e) {
        float v = acc[i][j][e] * alpha + bsv;
        if (gelu) v = 0.5f * v * (1.0f + erff(v * 0.70710678118654752f));
        vals[e] = v;
      }
      if (Cf) {
        #pragma unroll
        for (int e = 0; e < 8; ++e)
          Cf[cOff + (long long)(rb + i * 16 + e) * ldc + col] = vals[e];
      }
      if (Cb) {
        #pragma unroll
        for (int e = 0; e < 8; ++e)
          Cb[cOff + (long long)(rb + i * 16 + e) * ldc + col] = (bf16_t)vals[e];
      }
      if (Ct) {  // transposed store: 8 consecutive rows per lane -> one 16B store
        v8bf pk;
        #pragma unroll
        for (int e = 0; e < 8; ++e) pk[e] = (bf16_t)vals[e];
        *(v8bf*)&Ct[ctOff + (long long)col * ldct + (rb + i * 16)] = pk;
      }
    }
  }
}

// ---------------------------------------------------------------------------
// Row softmax: 1 block per row, 4096 cols (16 per thread, held in registers)
// ---------------------------------------------------------------------------
__launch_bounds__(256)
__global__ void k_softmax(const float* __restrict__ S, bf16_t* __restrict__ P, int cols) {
  __shared__ float red[256];
  const int row = blockIdx.x;
  const int t = threadIdx.x;
  const float* srow = S + (long long)row * cols;
  float v[16];
  float mx = -3.4e38f;
  #pragma unroll
  for (int i = 0; i < 16; ++i) {
    v[i] = srow[i * 256 + t];
    mx = fmaxf(mx, v[i]);
  }
  red[t] = mx; __syncthreads();
  for (int s = 128; s > 0; s >>= 1) { if (t < s) red[t] = fmaxf(red[t], red[t + s]); __syncthreads(); }
  mx = red[0]; __syncthreads();
  float sum = 0.0f;
  #pragma unroll
  for (int i = 0; i < 16; ++i) { v[i] = __expf(v[i] - mx); sum += v[i]; }
  red[t] = sum; __syncthreads();
  for (int s = 128; s > 0; s >>= 1) { if (t < s) red[t] += red[t + s]; __syncthreads(); }
  const float inv = 1.0f / red[0];
  bf16_t* prow = P + (long long)row * cols;
  #pragma unroll
  for (int i = 0; i < 16; ++i) prow[i * 256 + t] = (bf16_t)(v[i] * inv);
}

// ---------------------------------------------------------------------------
// y = LayerNorm(a + b) * gamma + beta  (D=1024: 4 elems/thread, LDS reductions)
// ---------------------------------------------------------------------------
__launch_bounds__(256)
__global__ void k_res_ln(const float* __restrict__ a, const float* __restrict__ b,
                         const float* __restrict__ gamma, const float* __restrict__ beta,
                         float* __restrict__ outf, bf16_t* __restrict__ outb, int D) {
  __shared__ float red[256];
  const int row = blockIdx.x;
  const int t = threadIdx.x;
  float x[4];
  float s = 0.0f;
  #pragma unroll
  for (int i = 0; i < 4; ++i) {
    const int c = i * 256 + t;
    x[i] = a[(long long)row * D + c] + b[(long long)row * D + c];
    s += x[i];
  }
  red[t] = s; __syncthreads();
  for (int st = 128; st > 0; st >>= 1) { if (t < st) red[t] += red[t + st]; __syncthreads(); }
  const float mu = red[0] / (float)D; __syncthreads();
  float s2 = 0.0f;
  #pragma unroll
  for (int i = 0; i < 4; ++i) { const float d = x[i] - mu; s2 += d * d; }
  red[t] = s2; __syncthreads();
  for (int st = 128; st > 0; st >>= 1) { if (t < st) red[t] += red[t + st]; __syncthreads(); }
  const float inv = rsqrtf(red[0] / (float)D + 1e-5f);
  #pragma unroll
  for (int i = 0; i < 4; ++i) {
    const int c = i * 256 + t;
    const float y = (x[i] - mu) * inv * gamma[c] + beta[c];
    outf[(long long)row * D + c] = y;
    if (outb) outb[(long long)row * D + c] = (bf16_t)y;
  }
}

// ---------------------------------------------------------------------------
extern "C" void kernel_launch(void* const* d_in, const int* in_sizes, int n_in,
                              void* d_out, int out_size, void* d_ws, size_t ws_size,
                              hipStream_t stream) {
  (void)in_sizes; (void)n_in; (void)out_size; (void)ws_size;
  const int B = 2, S = 4096, D = 1024, H = 4096;
  const int M = B * S;  // 8192

  const float* X     = (const float*)d_in[0];
  const float* Wq    = (const float*)d_in[1];
  const float* bq    = (const float*)d_in[2];
  const float* Wk    = (const float*)d_in[3];
  const float* bk    = (const float*)d_in[4];
  const float* Wv    = (const float*)d_in[5];
  const float* bv    = (const float*)d_in[6];
  const float* W1    = (const float*)d_in[7];
  const float* b1    = (const float*)d_in[8];
  const float* W2    = (const float*)d_in[9];
  const float* b2    = (const float*)d_in[10];
  const float* gamma = (const float*)d_in[11];
  const float* beta  = (const float*)d_in[12];
  float* out = (float*)d_out;

  // --- workspace bump allocation (256B aligned) ---
  char* ws = (char*)d_ws;
  size_t off = 0;
  auto alloc = [&](size_t bytes) -> char* {
    char* p = ws + off;
    off += (bytes + 255) & ~(size_t)255;
    return p;
  };
  bf16_t* XB  = (bf16_t*)alloc((size_t)M * D * 2);   // X in bf16
  bf16_t* WQT = (bf16_t*)alloc((size_t)D * D * 2);   // Wq^T  [N=D][K=D]
  bf16_t* WKT = (bf16_t*)alloc((size_t)D * D * 2);
  bf16_t* WVT = (bf16_t*)alloc((size_t)D * D * 2);
  bf16_t* W1T = (bf16_t*)alloc((size_t)D * H * 2);   // W1^T  [N=H][K=D]
  bf16_t* W2T = (bf16_t*)alloc((size_t)H * D * 2);   // W2^T  [N=D][K=H]
  bf16_t* QPB = (bf16_t*)alloc((size_t)M * D * 2);   // q projection [M][D]
  bf16_t* KPB = (bf16_t*)alloc((size_t)M * D * 2);   // k projection [M][D]
  bf16_t* VPT = (bf16_t*)alloc((size_t)M * D * 2);   // v projection transposed: per batch [D][S]
  float*  SF  = (float*) alloc((size_t)M * S * 4);   // scores f32 (134MB)
  bf16_t* PB  = (bf16_t*)alloc((size_t)M * S * 2);   // probs bf16 (67MB)
  float*  AO  = (float*) alloc((size_t)M * D * 4);   // attention output f32
  // lifetime-based aliases:
  float*  XF  = (float*)PB;                                  // x f32 (P dead by then)
  bf16_t* XBF = (bf16_t*)((char*)PB + (size_t)M * D * 4);    // x bf16
  bf16_t* HB  = (bf16_t*)SF;                                 // h bf16 (S dead by then)
  float*  FF  = (float*)((char*)SF + (size_t)M * H * 2);     // ff f32

  // --- conversions (weights transposed to [N][K]) ---
  k_cvt_bf16<<<2048, 256, 0, stream>>>(X, XB, M * D);
  k_cvt_t_bf16<<<dim3(D / 32, D / 32), 256, 0, stream>>>(Wq, WQT, D, D);
  k_cvt_t_bf16<<<dim3(D / 32, D / 32), 256, 0, stream>>>(Wk, WKT, D, D);
  k_cvt_t_bf16<<<dim3(D / 32, D / 32), 256, 0, stream>>>(Wv, WVT, D, D);
  k_cvt_t_bf16<<<dim3(H / 32, D / 32), 256, 0, stream>>>(W1, W1T, D, H);
  k_cvt_t_bf16<<<dim3(D / 32, H / 32), 256, 0, stream>>>(W2, W2T, H, D);

  // --- Q/K projections: [8192,1024] = XB @ W^T^T + b, bf16 out ---
  dim3 gq(D / 128, M / 128, 1);
  k_gemm_wmma<<<gq, 256, 0, stream>>>(XB, 0, WQT, 0, bq,
                                      nullptr, QPB, 0, nullptr, 0, 0,
                                      D, D, D, D, 1.0f, 0);
  k_gemm_wmma<<<gq, 256, 0, stream>>>(XB, 0, WKT, 0, bk,
                                      nullptr, KPB, 0, nullptr, 0, 0,
                                      D, D, D, D, 1.0f, 0);
  // --- V projection, batched, transposed output VPT[b][d][s] ---
  dim3 gv(D / 128, S / 128, B);
  k_gemm_wmma<<<gv, 256, 0, stream>>>(XB, (long long)S * D, WVT, 0, bv,
                                      nullptr, nullptr, 0, VPT, (long long)D * S, S,
                                      D, D, D, D, 1.0f, 0);

  // --- scores = (Kp @ Qp^T) / sqrt(D), per batch (faithful Q<-k quirk) ---
  dim3 gs(S / 128, S / 128, B);
  k_gemm_wmma<<<gs, 256, 0, stream>>>(KPB, (long long)S * D, QPB, (long long)S * D, nullptr,
                                      SF, nullptr, (long long)S * S, nullptr, 0, 0,
                                      D, D, D, S, 0.03125f, 0);

  // --- softmax rows -> bf16 probs ---
  k_softmax<<<M, 256, 0, stream>>>(SF, PB, S);

  // --- attn_out = P @ Vp  (B operand = VPT, already [N=D][K=S]) ---
  dim3 ga(D / 128, S / 128, B);
  k_gemm_wmma<<<ga, 256, 0, stream>>>(PB, (long long)S * S, VPT, (long long)D * S, nullptr,
                                      AO, nullptr, (long long)S * D, nullptr, 0, 0,
                                      S, S, S, D, 1.0f, 0);

  // --- x = LN(input + attn_out), keep f32 + bf16 copies ---
  k_res_ln<<<M, 256, 0, stream>>>(X, AO, gamma, beta, XF, XBF, D);

  // --- FFN1: h = gelu(x @ W1 + b1), bf16 out ---
  dim3 g1(H / 128, M / 128, 1);
  k_gemm_wmma<<<g1, 256, 0, stream>>>(XBF, 0, W1T, 0, b1,
                                      nullptr, HB, 0, nullptr, 0, 0,
                                      D, D, D, H, 1.0f, /*gelu=*/1);

  // --- FFN2: ff = h @ W2 + b2, f32 out ---
  dim3 g2(D / 128, M / 128, 1);
  k_gemm_wmma<<<g2, 256, 0, stream>>>(HB, 0, W2T, 0, b2,
                                      FF, nullptr, 0, nullptr, 0, 0,
                                      H, H, H, D, 1.0f, 0);

  // --- out = LN(x + ff) ---
  k_res_ln<<<M, 256, 0, stream>>>(XF, FF, gamma, beta, out, nullptr, D);
}
